// Sty_layer_6390911336533
// MI455X (gfx1250) — matrix operand order repaired and checked by reading, not compiled, not run on hardware
//
#include <hip/hip_runtime.h>
#include <hip/hip_bf16.h>

typedef __attribute__((ext_vector_type(16))) _Float16 v16h;
typedef __attribute__((ext_vector_type(8)))  _Float16 v8h;
typedef __attribute__((ext_vector_type(2)))  _Float16 h2;
typedef __attribute__((ext_vector_type(8)))  float    v8f;

#define BB   8
#define CI   256
#define CO   64
#define HID  16
#define HH   256
#define WW   256
#define KK9  9
#define KDIM 576          // CO * 9
#define WPAD 584          // padded weight row (f16): 1168B = 73*16B (odd multiple of 16B)
#define CPAD 72           // padded channel dim in tile (f16): 144B = 9*16B (odd multiple)
#define TILE 128          // pixels per row segment
#define ROWS 2            // output rows per block

union FragU { v16h v; v8h h[2]; };

// ---------------------------------------------------------------------------
// Kernel 1: style MLP + demod scalar + reordered, PADDED f16 weights
// grid = B*CO, 256 threads. w16[b][o][WPAD], kidx=(kh*3+kw)*64+c
// ---------------------------------------------------------------------------
__global__ __launch_bounds__(256)
void style_weights_kernel(const float* __restrict__ vec, const float* __restrict__ w1,
                          const float* __restrict__ w2, const float* __restrict__ wconv,
                          _Float16* __restrict__ w16) {
    int b = blockIdx.x >> 6;
    int o = blockIdx.x & 63;
    int t = threadIdx.x;
    __shared__ float hid[HID];
    __shared__ float red[256];
    __shared__ float a_sh;

    if (t < HID) {
        float s = 0.f;
        const float* vp = vec + b * CI;
        const float* wp = w1 + t * CI;
        for (int c = 0; c < CI; ++c) s += vp[c] * wp[c];
        hid[t] = s > 0.f ? s : 0.1f * s;             // LeakyReLU(0.1)
    }
    __syncthreads();

    float ss = 0.f;
    const float* wc = wconv + o * KDIM;
    for (int i = t; i < KDIM; i += 256) { float w = wc[i]; ss += w * w; }
    red[t] = ss;
    __syncthreads();
    for (int s2 = 128; s2 > 0; s2 >>= 1) {
        if (t < s2) red[t] += red[t + s2];
        __syncthreads();
    }
    if (t == 0) {
        float sty = 0.f;
        const float* w2p = w2 + o * HID;
        for (int r = 0; r < HID; ++r) sty += hid[r] * w2p[r];
        float s1 = sty + 1.0f;
        a_sh = s1 * rsqrtf(s1 * s1 * red[0] + 1e-8f);
    }
    __syncthreads();
    float a = a_sh;

    _Float16* dst = w16 + ((size_t)(b * CO + o)) * WPAD;
    for (int i = t; i < KDIM; i += 256) {
        int kk = i >> 6;
        int c  = i & 63;
        dst[i] = (_Float16)(wc[c * KK9 + kk] * a);
    }
    if (t < WPAD - KDIM) dst[KDIM + t] = (_Float16)0.f;
}

// ---------------------------------------------------------------------------
// Kernel 2: ChanNorm, single pass. 4 lanes/pixel x 16 channels, shfl_xor reduce.
// ---------------------------------------------------------------------------
__global__ __launch_bounds__(256)
void channorm_kernel(const float* __restrict__ x, const float* __restrict__ g,
                     const float* __restrict__ bch, _Float16* __restrict__ nx) {
    size_t gid   = (size_t)blockIdx.x * 256 + threadIdx.x;
    size_t pixel = gid >> 2;
    int    cg    = (int)(gid & 3);
    size_t b     = pixel >> 16;
    size_t pos   = pixel & 65535;
    size_t base  = ((b * CO) << 16) + pos + ((size_t)(cg * 16) << 16);

    float v[16];
    float sum = 0.f, sq = 0.f;
    #pragma unroll
    for (int j = 0; j < 16; ++j) {
        v[j] = x[base + ((size_t)j << 16)];
        sum += v[j]; sq += v[j] * v[j];
    }
    sum += __shfl_xor(sum, 1); sq += __shfl_xor(sq, 1);
    sum += __shfl_xor(sum, 2); sq += __shfl_xor(sq, 2);

    float mean = sum * (1.0f / CO);
    float var  = sq * (1.0f / CO) - mean * mean;
    float inv  = rsqrtf(var + 1e-5f);
    #pragma unroll
    for (int j = 0; j < 16; ++j) {
        int c = cg * 16 + j;
        nx[base + ((size_t)j << 16)] = (_Float16)((v[j] - mean) * inv * g[c] + bch[c]);
    }
}

// ---------------------------------------------------------------------------
// Kernel 3: WMMA conv. Block tile: 64(M) x 2 rows x 128 px. 8 waves; wave tile
// 32(M) x 32(N) x 2 rows = 8 accumulators. A-frags shared across rows.
// grid = B * (H/2) * (W/TILE) = 2048
// ---------------------------------------------------------------------------
__global__ __launch_bounds__(256)
void conv_wmma_kernel(const _Float16* __restrict__ nx, const _Float16* __restrict__ w16,
                      const float* __restrict__ x, float* __restrict__ out) {
    __shared__ __align__(16) _Float16 sW[CO][WPAD];                 // 74,752 B
    __shared__ __align__(16) _Float16 sX[ROWS + 2][TILE + 2][CPAD]; // 74,880 B

    int bi = blockIdx.x;
    int b  = bi >> 8;                  // 256 blocks per sample
    int r8 = bi & 255;
    int h0 = (r8 >> 1) << 1;           // first of two output rows
    int w0 = (r8 & 1) * TILE;
    int t  = threadIdx.x;

    // --- async-stage padded weights: 4672 x 16B chunks, uniform trip count
    {
        unsigned ldsW = (unsigned)(unsigned long long)&sW[0][0];
        const _Float16* gsrc = w16 + (size_t)b * CO * WPAD;
        #pragma unroll 1
        for (int k = 0; k < 18; ++k) {
            unsigned off = (unsigned)(t + (k << 8)) * 16u;
            asm volatile("global_load_async_to_lds_b128 %0, %1, %2"
                         :: "v"(ldsW + off), "v"(off), "s"(gsrc) : "memory");
        }
        if (t < 64) {
            unsigned off = (unsigned)(t + 4608) * 16u;
            asm volatile("global_load_async_to_lds_b128 %0, %1, %2"
                         :: "v"(ldsW + off), "v"(off), "s"(gsrc) : "memory");
        }
    }

    // --- stage input tile transposed to [row][col][ch], paired-channel b32 stores
    {
        const _Float16* gx = nx + (((size_t)b * CO) << 16);
        // interior: 4 rows x 32 ch-pairs x 16 col-chunks = 2048 -> 8/thread
        #pragma unroll 1
        for (int k = 0; k < 8; ++k) {
            int i    = t + (k << 8);
            int col8 = (i & 15) * 8;
            int c    = ((i >> 4) & 31) * 2;
            int r    = i >> 9;
            int hh   = h0 + r - 1;
            v8h va = (v8h)(_Float16)0.f, vb = va;
            if (hh >= 0 && hh < HH) {
                const _Float16* p = gx + (((size_t)c) << 16) + ((size_t)hh << 8) + w0 + col8;
                va = *(const v8h*)(p);
                vb = *(const v8h*)(p + 65536);
            }
            #pragma unroll
            for (int j = 0; j < 8; ++j) {
                h2 pr; pr.x = va[j]; pr.y = vb[j];
                *(h2*)&sX[r][col8 + 1 + j][c] = pr;
            }
        }
        // halo columns 0 and TILE+1: 4 rows x 64 ch x 2 sides = 512 -> 2/thread
        #pragma unroll 1
        for (int k = 0; k < 2; ++k) {
            int i    = t + (k << 8);
            int side = i & 1;
            int c    = (i >> 1) & 63;
            int r    = i >> 7;
            int hh   = h0 + r - 1;
            int wcol = side ? (w0 + TILE) : (w0 - 1);
            _Float16 v = (_Float16)0.f;
            if (hh >= 0 && hh < HH && wcol >= 0 && wcol < WW)
                v = gx[(((size_t)c) << 16) + ((size_t)hh << 8) + wcol];
            sX[r][side ? (TILE + 1) : 0][c] = v;
        }
    }
    asm volatile("s_wait_asynccnt 0x0" ::: "memory");
    __syncthreads();

    // --- compute: wave tile 2m x 2n x 2 rows
    int lane = t & 31;
    int ln   = lane & 15;
    int hi   = lane >> 4;
    int wave = t >> 5;
    int mh   = wave & 1;
    int nh   = wave >> 1;
    int np0  = nh * 32 + ln;

    v8f zero = {0.f, 0.f, 0.f, 0.f, 0.f, 0.f, 0.f, 0.f};
    v8f acc[ROWS][2][2];
    #pragma unroll
    for (int rr = 0; rr < ROWS; ++rr)
        #pragma unroll
        for (int mi = 0; mi < 2; ++mi)
            #pragma unroll
            for (int ni = 0; ni < 2; ++ni) acc[rr][mi][ni] = zero;

    const _Float16* wr0 = &sW[(2 * mh)     * 16 + ln][hi * 8];
    const _Float16* wr1 = &sW[(2 * mh + 1) * 16 + ln][hi * 8];

    #pragma unroll
    for (int kk = 0; kk < KK9; ++kk) {
        int kh = kk / 3, kw = kk % 3;
        #pragma unroll
        for (int kc = 0; kc < 2; ++kc) {
            int cb = kc * 32;
            int wo = kk * 64 + kc * 32;
            FragU fa0, fa1;
            fa0.h[0] = *(const v8h*)(wr0 + wo); fa0.h[1] = *(const v8h*)(wr0 + wo + 16);
            fa1.h[0] = *(const v8h*)(wr1 + wo); fa1.h[1] = *(const v8h*)(wr1 + wo + 16);
            #pragma unroll
            for (int rr = 0; rr < ROWS; ++rr) {
                const _Float16* x0 = &sX[kh + rr][np0 + kw][hi * 8];
                FragU fb0, fb1;
                fb0.h[0] = *(const v8h*)(x0 + cb);
                fb0.h[1] = *(const v8h*)(x0 + cb + 16);
                fb1.h[0] = *(const v8h*)(x0 + CPAD * 16 + cb);
                fb1.h[1] = *(const v8h*)(x0 + CPAD * 16 + cb + 16);
                acc[rr][0][0] = __builtin_amdgcn_wmma_f32_16x16x32_f16(false, fa0.v, false, fb0.v,
                                                   (short)0, acc[rr][0][0], false, false);
                acc[rr][0][1] = __builtin_amdgcn_wmma_f32_16x16x32_f16(false, fa0.v, false, fb1.v,
                                                   (short)0, acc[rr][0][1], false, false);
                acc[rr][1][0] = __builtin_amdgcn_wmma_f32_16x16x32_f16(false, fa1.v, false, fb0.v,
                                                   (short)0, acc[rr][1][0], false, false);
                acc[rr][1][1] = __builtin_amdgcn_wmma_f32_16x16x32_f16(false, fa1.v, false, fb1.v,
                                                   (short)0, acc[rr][1][1], false, false);
            }
        }
    }

    // --- writeout with fused residual. D: VGPR r -> M = 16*mt + r + 8*hi, N = ln
    size_t base0 = (((size_t)b * CO) << 16) + ((size_t)h0 << 8) + w0 + np0;
    #pragma unroll
    for (int rr = 0; rr < ROWS; ++rr) {
        #pragma unroll
        for (int mi = 0; mi < 2; ++mi) {
            int ob = (2 * mh + mi) * 16 + hi * 8;
            #pragma unroll
            for (int r = 0; r < 8; ++r) {
                size_t i0 = base0 + ((size_t)(ob + r) << 16) + (rr << 8);
                size_t i1 = i0 + 16;
                out[i0] = acc[rr][mi][0][r] + x[i0];
                out[i1] = acc[rr][mi][1][r] + x[i1];
            }
        }
    }
}

// ---------------------------------------------------------------------------
extern "C" void kernel_launch(void* const* d_in, const int* in_sizes, int n_in,
                              void* d_out, int out_size, void* d_ws, size_t ws_size,
                              hipStream_t stream) {
    const float* x     = (const float*)d_in[0];
    const float* vec   = (const float*)d_in[1];
    const float* g     = (const float*)d_in[2];
    const float* bch   = (const float*)d_in[3];
    const float* w1    = (const float*)d_in[4];
    const float* w2    = (const float*)d_in[5];
    const float* wconv = (const float*)d_in[6];
    float* out = (float*)d_out;

    _Float16* nx  = (_Float16*)d_ws;
    _Float16* w16 = (_Float16*)((char*)d_ws + (size_t)BB * CO * HH * WW * sizeof(_Float16));

    hipLaunchKernelGGL(style_weights_kernel, dim3(BB * CO), dim3(256), 0, stream,
                       vec, w1, w2, wconv, w16);
    hipLaunchKernelGGL(channorm_kernel, dim3((BB * HH * WW * 4) / 256), dim3(256), 0, stream,
                       x, g, bch, nx);
    hipLaunchKernelGGL(conv_wmma_kernel, dim3(BB * (HH / ROWS) * (WW / TILE)), dim3(256), 0, stream,
                       nx, w16, x, out);
}
